// DepressionQuestionModel_36953898615395
// MI455X (gfx1250) — matrix operand order repaired and checked by reading, not compiled, not run on hardware
//
#include <hip/hip_runtime.h>
#include <hip/hip_bf16.h>
#include <math.h>

// ---------------- problem constants ----------------
#define NB    64              // batch
#define NS    9               // symptoms
#define NQ    16              // questions per symptom
#define NT    208             // NB + NS*NQ samples through branch_feats
#define CIN   128             // conv1 in-channels
#define LIN   768             // conv1 input length
#define H1    256             // conv1 out-channels
#define KW1   5
#define KDIM  (CIN*KW1)       // 640 GEMM-K
#define LA    382             // after pool
#define LA_PAD 384
#define LA2   191             // after second pool
#define LA2_PAD 192
#define D5    189
#define D2    95
#define D3    93
#define E5P   192
#define E2P   96
#define E3P   96

#define NCHUNK 20             // K chunks of 32 (kk-major: K = kk*128 + ci)
#define XCOLS  36             // 32 outputs + 4 halo
#define XPITCH 136            // LDS col stride in halves (bank-friendly, 16B aligned)

// ---------------- bf16 helpers (bit-level) ----------------
__device__ __forceinline__ unsigned short f2bf(float f) {
    unsigned u = __float_as_uint(f);
    u += 0x7FFFu + ((u >> 16) & 1u);     // round-to-nearest-even
    return (unsigned short)(u >> 16);
}
__device__ __forceinline__ float bf2f(unsigned short h) {
    return __uint_as_float(((unsigned)h) << 16);
}

// ---------------- WMMA vector types ----------------
typedef __attribute__((ext_vector_type(16))) __bf16 v16bf;
typedef __attribute__((ext_vector_type(8)))  __bf16 v8bf;
typedef __attribute__((ext_vector_type(8)))  float  v8f;

union BF16x16 { v16bf v; v8bf h[2]; unsigned short s[16]; };

// =====================================================================
// K0: repack conv1 weights into per-lane WMMA A-fragment order (bf16).
//   Fragment (ct, c, lane) holds 16 halves e=0..15:
//     m = lane&15, kg = lane>>4
//     koff = kg*16+e                             (contiguous K order)
//     kk = c>>2, ci = (c&3)*32 + koff            (kk-major K order)
//     value = w1[(ct*16+m)*640 + ci*5 + kk]
// =====================================================================
__global__ __launch_bounds__(256) void repack_w1_kernel(
    const float* __restrict__ w1, unsigned short* __restrict__ aw)
{
    const int idx = blockIdx.x * 256 + threadIdx.x;   // 16*20*32*16 = 163840
    const int e    = idx & 15;
    const int lid  = (idx >> 4) & 31;
    const int rest = idx >> 9;          // 0..319
    const int c    = rest % NCHUNK;
    const int ct   = rest / NCHUNK;
    const int m    = lid & 15;
    const int kg   = lid >> 4;
    const int koff = kg * 16 + e;
    const int kk   = c >> 2;
    const int ci   = (c & 3) * 32 + koff;
    aw[idx] = f2bf(w1[(size_t)(ct * 16 + m) * KDIM + ci * KW1 + kk]);
}

// =====================================================================
// K1: conv1 implicit GEMM via v_wmma_f32_16x16x32_bf16 + fused maxpool2.
//   Block = 256 thr (8 waves), all waves share one x tile (LDS-staged).
//   Per wave: 16(co) x 32(l) raw tile -> 16x16 pooled tile of `a`.
// =====================================================================
__global__ __launch_bounds__(256) void conv1_wmma_kernel(
    const float* __restrict__ in1, const float* __restrict__ qst,
    const unsigned short* __restrict__ aw, const float* __restrict__ b1,
    unsigned short* __restrict__ a_out)
{
    __shared__ unsigned short xs[XCOLS * XPITCH];     // xs[col][ci], bf16

    const int tid  = threadIdx.x;
    const int lane = tid & 31;
    const int wave = tid >> 5;
    const int t    = blockIdx.x / 48;
    const int blk  = blockIdx.x % 48;
    const int lp_tile = blk >> 1;                     // 0..23 (shared per block)
    const int co_tile = (blk & 1) * 8 + wave;         // 0..15
    const int co0 = co_tile * 16;
    const int l0  = lp_tile * 32;

    const float* __restrict__ xb = (t < NB)
        ? (in1 + (size_t)t * CIN * LIN)
        : (qst + (size_t)(t - NB) * CIN * LIN);

    // ---- cooperative stage: x[ci][l0..l0+35] -> xs[col][ci] (transposed bf16)
    for (int idx = tid; idx < CIN * XCOLS; idx += 256) {
        const int ci = idx / XCOLS;
        const int j  = idx - ci * XCOLS;
        const int l  = l0 + j;
        const float v = (l < LIN) ? xb[(size_t)ci * LIN + l] : 0.f;
        xs[j * XPITCH + ci] = f2bf(v);
    }
    __syncthreads();

    const int mn = lane & 15;       // A row M / B,C col N
    const int kg = lane >> 4;       // lane-half: K sub-range select

    v8f acc0 = {};                  // columns l0+0..15
    v8f acc1 = {};                  // columns l0+16..31

    const unsigned short* __restrict__ awp =
        aw + ((size_t)co_tile * NCHUNK * 32 + lane) * 16;

    for (int c = 0; c < NCHUNK; ++c) {
        const int kk  = c >> 2;                    // constant per chunk
        const int ci0 = (c & 3) * 32 + kg * 16;    // contiguous ci run per lane

        BF16x16 A, B0, B1;
        // A fragment: one coalesced 32B read per lane (2x b128)
        const unsigned short* ap = awp + c * (32 * 16);
        A.h[0] = *(const v8bf*)(ap);
        A.h[1] = *(const v8bf*)(ap + 8);
        if (c + 1 < NCHUNK)
            __builtin_prefetch(awp + (c + 1) * (32 * 16), 0, 1);

        // B fragments: 16 contiguous bf16 in LDS (2x ds_load_b128 each)
        const unsigned short* p0 = xs + (mn + kk) * XPITCH + ci0;
        const unsigned short* p1 = xs + (mn + 16 + kk) * XPITCH + ci0;
        B0.h[0] = *(const v8bf*)(p0);
        B0.h[1] = *(const v8bf*)(p0 + 8);
        B1.h[0] = *(const v8bf*)(p1);
        B1.h[1] = *(const v8bf*)(p1 + 8);

        acc0 = __builtin_amdgcn_wmma_f32_16x16x32_bf16(
                   false, A.v, false, B0.v, (short)0, acc0, false, false);
        acc1 = __builtin_amdgcn_wmma_f32_16x16x32_bf16(
                   false, A.v, false, B1.v, (short)0, acc1, false, false);
    }

    // ---- fused MaxPool1d(2): adjacent l live in adjacent lanes ----
    const int  lp0    = lp_tile * 16;
    const bool writer = ((mn & 1) == 0);
    #pragma unroll
    for (int r = 0; r < 8; ++r) {
        float p0 = fmaxf(acc0[r], __shfl_xor(acc0[r], 1, 32));
        float p1 = fmaxf(acc1[r], __shfl_xor(acc1[r], 1, 32));
        if (writer) {
            const int m  = r + 8 * kg;            // C layout: VGPR r -> M=r / M=r+8
            const int co = co0 + m;
            const float bias = b1[co];            // bias commutes with max
            const size_t base = ((size_t)t * H1 + co) * LA_PAD;
            const int p = lp0 + (mn >> 1);
            a_out[base + p]     = f2bf(p0 + bias);
            a_out[base + p + 8] = f2bf(p1 + bias);
        }
    }
}

// =====================================================================
// K2: a2 = maxpool2(a)
// =====================================================================
__global__ void pool2_kernel(const unsigned short* __restrict__ a,
                             unsigned short* __restrict__ a2)
{
    int idx = blockIdx.x * blockDim.x + threadIdx.x;
    const int total = NT * H1 * LA2;
    if (idx >= total) return;
    const int p  = idx % LA2;
    const int tc = idx / LA2;
    const size_t src = (size_t)tc * LA_PAD + 2 * p;
    const float v = fmaxf(bf2f(a[src]), bf2f(a[src + 1]));
    a2[(size_t)tc * LA2_PAD + p] = f2bf(v);
}

// =====================================================================
// K3: conv2 branches -> e5 / e2 / e3 (weights staged in LDS).
//   Replicates reference bug: e3 uses conv2_w / conv2_b on a2.
// =====================================================================
__global__ __launch_bounds__(256) void conv2_kernel(
    const unsigned short* __restrict__ a, const unsigned short* __restrict__ a2,
    const float* __restrict__ w2,  const float* __restrict__ b2,
    const float* __restrict__ w22, const float* __restrict__ b22,
    float* __restrict__ e5, float* __restrict__ e2, float* __restrict__ e3)
{
    __shared__ float sw5[H1 * 5];
    __shared__ float sw2[H1 * 2];
    const int t = blockIdx.x;
    for (int i = threadIdx.x; i < H1 * 5; i += 256) sw5[i] = w2[i];
    for (int i = threadIdx.x; i < H1 * 2; i += 256) sw2[i] = w22[i];
    __syncthreads();

    const unsigned short* At  = a  + (size_t)t * H1 * LA_PAD;
    const unsigned short* A2t = a2 + (size_t)t * H1 * LA2_PAD;
    const int j = threadIdx.x;

    if (j < D5) {
        float s0 = 0.f, s1 = 0.f;
        for (int c = 0; c < H1; ++c) {
            const unsigned short* row = At + (size_t)c * LA_PAD + 2 * j;
            #pragma unroll
            for (int k = 0; k < 5; ++k) {
                const float w = sw5[c * 5 + k];
                s0 += w * bf2f(row[k]);
                s1 += w * bf2f(row[k + 1]);
            }
        }
        e5[t * E5P + j] = fmaxf(s0, s1) + b2[0];
    }
    if (j < D2) {
        float s0 = 0.f, s1 = 0.f;
        for (int c = 0; c < H1; ++c) {
            const unsigned short* row = A2t + (size_t)c * LA2_PAD + 2 * j;
            #pragma unroll
            for (int k = 0; k < 2; ++k) {
                const float w = sw2[c * 2 + k];
                s0 += w * bf2f(row[k]);
                s1 += w * bf2f(row[k + 1]);
            }
        }
        e2[t * E2P + j] = fmaxf(s0, s1) + b22[0];
    }
    if (j < D3) {
        float s0 = 0.f, s1 = 0.f;
        for (int c = 0; c < H1; ++c) {
            const unsigned short* row = A2t + (size_t)c * LA2_PAD + 2 * j;
            #pragma unroll
            for (int k = 0; k < 5; ++k) {
                const float w = sw5[c * 5 + k];
                s0 += w * bf2f(row[k]);
                s1 += w * bf2f(row[k + 1]);
            }
        }
        e3[t * E3P + j] = fmaxf(s0, s1) + b2[0];
    }
}

// =====================================================================
// K4: cosine means -> distance [64,9]; logits -> softmax -> out [64,2]
// =====================================================================
__device__ __forceinline__ float cos_pair(const float* __restrict__ x,
                                          const float* __restrict__ y, int D)
{
    float dot = 0.f, nx = 0.f, ny = 0.f;
    for (int d = 0; d < D; ++d) {
        const float a = x[d], b = y[d];
        dot += a * b; nx += a * a; ny += b * b;
    }
    return dot / fmaxf(sqrtf(nx) * sqrtf(ny), 1e-8f);
}

__global__ __launch_bounds__(192) void cosine_kernel(
    const float* __restrict__ e5, const float* __restrict__ e2,
    const float* __restrict__ e3,
    const float* __restrict__ symw, const float* __restrict__ symb,
    float* __restrict__ out)
{
    __shared__ float scos[NS * NQ];
    __shared__ float sdist[NS];
    const int b = blockIdx.x;
    const int tid = threadIdx.x;

    if (tid < NS * NQ) {
        const int qrow = NB + tid;
        float tot = 0.f;
        tot += cos_pair(e5 + b * E5P, e5 + qrow * E5P, D5);
        tot += cos_pair(e2 + b * E2P, e2 + qrow * E2P, D2);
        tot += cos_pair(e3 + b * E3P, e3 + qrow * E3P, D3);
        scos[tid] = tot;
    }
    __syncthreads();
    if (tid < NS) {
        float s = 0.f;
        for (int q = 0; q < NQ; ++q) s += scos[tid * NQ + q];
        const float d = s / (3.0f * NQ);
        sdist[tid] = d;
        out[1 + NB * 2 + b * NS + tid] = d;
    }
    __syncthreads();
    if (tid == 0) {
        float l0 = symb[0], l1 = symb[1];
        for (int s = 0; s < NS; ++s) {
            l0 += sdist[s] * symw[s];
            l1 += sdist[s] * symw[NS + s];
        }
        const float m  = fmaxf(l0, l1);
        const float z0 = __expf(l0 - m), z1 = __expf(l1 - m);
        const float inv = 1.f / (z0 + z1);
        out[1 + b * 2 + 0] = z0 * inv;
        out[1 + b * 2 + 1] = z1 * inv;
    }
}

// =====================================================================
// K5: loss = CE applied to softmax probabilities (reference bug kept)
// =====================================================================
__global__ void loss_kernel(const int* __restrict__ labels, float* __restrict__ out)
{
    __shared__ float red[NB];
    const int b = threadIdx.x;
    const float p0 = out[1 + b * 2], p1 = out[2 + b * 2];
    const float m   = fmaxf(p0, p1);
    const float lse = m + logf(__expf(p0 - m) + __expf(p1 - m));
    const int lab = labels[b];
    red[b] = ((lab == 0) ? p0 : p1) - lse;
    __syncthreads();
    for (int off = NB / 2; off > 0; off >>= 1) {
        if (b < off) red[b] += red[b + off];
        __syncthreads();
    }
    if (b == 0) out[0] = -red[0] / (float)NB;
}

// =====================================================================
// Launcher
// =====================================================================
extern "C" void kernel_launch(void* const* d_in, const int* in_sizes, int n_in,
                              void* d_out, int out_size, void* d_ws, size_t ws_size,
                              hipStream_t stream)
{
    const float* in1   = (const float*)d_in[0];
    const float* qst   = (const float*)d_in[1];
    const int*   labels= (const int*)  d_in[2];
    const float* w1    = (const float*)d_in[3];
    const float* b1    = (const float*)d_in[4];
    const float* w2    = (const float*)d_in[5];
    const float* b2    = (const float*)d_in[6];
    const float* w22   = (const float*)d_in[7];
    const float* b22   = (const float*)d_in[8];
    const float* symw  = (const float*)d_in[9];
    const float* symb  = (const float*)d_in[10];
    float* out = (float*)d_out;                   // 705 floats

    // workspace layout (bytes)
    const size_t SZ_A   = (size_t)NT * H1 * LA_PAD * 2;
    const size_t SZ_A2  = (size_t)NT * H1 * LA2_PAD * 2;
    const size_t SZ_E5  = (size_t)NT * E5P * 4;
    const size_t SZ_E2  = (size_t)NT * E2P * 4;
    const size_t SZ_E3  = (size_t)NT * E3P * 4;
    const size_t OFF_A  = 0;
    const size_t OFF_A2 = OFF_A  + SZ_A;
    const size_t OFF_E5 = OFF_A2 + SZ_A2;
    const size_t OFF_E2 = OFF_E5 + SZ_E5;
    const size_t OFF_E3 = OFF_E2 + SZ_E2;
    const size_t OFF_AW = OFF_E3 + SZ_E3;         // 16*20*32*16 bf16 = 320KB

    char* ws = (char*)d_ws;
    unsigned short* a  = (unsigned short*)(ws + OFF_A);
    unsigned short* a2 = (unsigned short*)(ws + OFF_A2);
    float* e5 = (float*)(ws + OFF_E5);
    float* e2 = (float*)(ws + OFF_E2);
    float* e3 = (float*)(ws + OFF_E3);
    unsigned short* aw = (unsigned short*)(ws + OFF_AW);

    repack_w1_kernel<<<(16 * NCHUNK * 32 * 16) / 256, 256, 0, stream>>>(w1, aw);

    conv1_wmma_kernel<<<NT * 48, 256, 0, stream>>>(in1, qst, aw, b1, a);

    const int tot2 = NT * H1 * LA2;
    pool2_kernel<<<(tot2 + 255) / 256, 256, 0, stream>>>(a, a2);

    conv2_kernel<<<NT, 256, 0, stream>>>(a, a2, w2, b2, w22, b22, e5, e2, e3);

    cosine_kernel<<<NB, 192, 0, stream>>>(e5, e2, e3, symw, symb, out);

    loss_kernel<<<1, NB, 0, stream>>>(labels, out);
}
